// DeformConv2D_27462020891385
// MI455X (gfx1250) — matrix operand order repaired and checked by reading, not compiled, not run on hardware
//
#include <hip/hip_runtime.h>

// CDNA5 / gfx1250 fused deformable-conv:
//   bilinear im2col (bf16) staged in LDS + v_wmma_f32_16x16x32_bf16 GEMM.
//   Weights pre-converted to bf16 once into d_ws; B tiles streamed into LDS
//   with GLOBAL_LOAD_ASYNC_TO_LDS_B128 (ASYNCcnt) when the builtin exists.
// M = 4*64*64 = 16384 output pixels, K = 128*9 = 1152, N = 256 out channels.

typedef __attribute__((ext_vector_type(16))) __bf16 v16bf;
typedef __attribute__((ext_vector_type(8)))  float  v8f;
typedef int v4i __attribute__((vector_size(16)));   // GCC vector: matches builtin param

#define BM   64          // output rows per workgroup
#define BK   32          // K chunk (== WMMA K)
#define KTOT 1152        // 128 channels * 9 taps
#define CIN  128
#define OC   256
#define LDA  40          // LDS row stride in bf16 (32 data + 8 pad -> 80B, bank-friendly)

#if defined(__gfx1250__) && \
    __has_builtin(__builtin_amdgcn_global_load_async_to_lds_b128) && \
    __has_builtin(__builtin_amdgcn_s_wait_asynccnt)
#define HAVE_ASYNC_LDS 1
#else
#define HAVE_ASYNC_LDS 0
#endif

__device__ __forceinline__ unsigned short f32_to_bf16(float f) {
  unsigned int u = __float_as_uint(f);
  unsigned int r = 0x7FFFu + ((u >> 16) & 1u);   // round-to-nearest-even
  return (unsigned short)((u + r) >> 16);
}

// ---- one-shot weight conversion: (256,128,3,3) f32 -> bf16 row-major [oc][k] ----
__global__ __launch_bounds__(256)
void cvt_weights_bf16(const float* __restrict__ w, unsigned short* __restrict__ wb) {
  int i = blockIdx.x * 256 + threadIdx.x;        // 0 .. 294911
  wb[i] = f32_to_bf16(w[i]);
}

template <bool WBF>
__global__ __launch_bounds__(256)
void DeformConv2D_wmma_kernel(const float* __restrict__ x,            // (4,128,64,64)
                              const float* __restrict__ off,          // (4,18,64,64)
                              const float* __restrict__ wgt,          // (256,128,3,3) f32
                              const unsigned short* __restrict__ wbf, // bf16 weights or null
                              float* __restrict__ out)                // (4,256,64,64)
{
  __shared__ int   s_idx[BM * 9 * 4];                 // gather base index per corner
  __shared__ float s_wgt[BM * 9 * 4];                 // bilinear weight (0 if OOB)
  __shared__ __align__(16) unsigned short As[BM * LDA];
  __shared__ __align__(16) unsigned short Bs[OC * LDA];

  const int tid = threadIdx.x;
  const int m0  = blockIdx.x * BM;

  // ---- 1. per-workgroup bilinear descriptors: 64 rows x 9 taps x 4 corners ----
  for (int e = tid; e < BM * 9; e += 256) {
    int ml = e / 9, n = e - ml * 9;
    int m  = m0 + ml;
    int b  = m >> 12;            // /4096
    int sp = m & 4095;
    int hh = sp >> 6, ww = sp & 63;

    float offx = off[((b * 18 + 2 * n    ) * 64 + hh) * 64 + ww];
    float offy = off[((b * 18 + 2 * n + 1) * 64 + hh) * 64 + ww];
    // padded-image coords (pad=1, padded size 66x66)
    float px = (float)(hh + 1) + (float)(n / 3 - 1) + offx;
    float py = (float)(ww + 1) + (float)(n % 3 - 1) + offy;
    float fx = floorf(px), fy = floorf(py);
    int qltx = min(max((int)fx, 0), 65);
    int qlty = min(max((int)fy, 0), 65);
    int qrbx = min(max((int)fx + 1, 0), 65);
    int qrby = min(max((int)fy + 1, 0), 65);
    // out-of-range snap (zero-grad path), then clamp — matches reference
    float pxs = (px < 1.f || px > 64.f) ? fx : px;
    float pys = (py < 1.f || py > 64.f) ? fy : py;
    pxs = fminf(fmaxf(pxs, 0.f), 65.f);
    pys = fminf(fmaxf(pys, 0.f), 65.f);
    float glx = 1.f + ((float)qltx - pxs);
    float grx = 1.f - ((float)qrbx - pxs);
    float gly = 1.f + ((float)qlty - pys);
    float gry = 1.f - ((float)qrby - pys);

    int bbase = b * (CIN * 4096);
    int qq[4][2] = {{qltx,qlty},{qrbx,qrby},{qltx,qrby},{qrbx,qlty}};
    float gg[4]  = {glx*gly,    grx*gry,    glx*gry,    grx*gly};
    int e4 = e * 4;
#pragma unroll
    for (int j = 0; j < 4; ++j) {
      int qx = qq[j][0], qy = qq[j][1];
      bool inb = (qx >= 1) & (qx <= 64) & (qy >= 1) & (qy <= 64);
      s_idx[e4 + j] = inb ? (bbase + (qx - 1) * 64 + (qy - 1)) : 0;
      s_wgt[e4 + j] = inb ? gg[j] : 0.f;                // zero-pad folded into weight
    }
  }
  __syncthreads();

  // ---- 2. wave tiling: wave owns 16 M-rows x 128 N-cols (8 WMMA tiles) ----
  const int wave = tid >> 5;
  const int lane = tid & 31;
  const int half = lane >> 4;         // lane group (ISA 16-bit VGPR layout)
  const int lm   = lane & 15;
  const int mrow = (wave & 3) * 16;   // tile-local M base
  const int ngrp = (wave >> 2) * 8;   // first N-subtile

  v8f acc[8];
#pragma unroll
  for (int i = 0; i < 8; ++i)
#pragma unroll
    for (int j = 0; j < 8; ++j) acc[i][j] = 0.f;

  for (int kc = 0; kc < KTOT; kc += BK) {
    // ---- stage B: 32(K) x 256(N) weight tile, stored [n][k] bf16 ----
    if (WBF) {
#if HAVE_ASYNC_LDS
      // async DMA: bf16 weights -> LDS, no VGPR round-trip (ASYNCcnt-tracked)
#pragma unroll
      for (int i = 0; i < 4; ++i) {
        int pid = i * 256 + tid;          // 0..1023, 16B each
        int nc = pid >> 2, q = pid & 3;
        v4i* gpv = reinterpret_cast<v4i*>(
            const_cast<unsigned short*>(wbf + nc * KTOT + kc + q * 8));
        v4i* lpv = reinterpret_cast<v4i*>(&Bs[nc * LDA + q * 8]);
        __builtin_amdgcn_global_load_async_to_lds_b128(
            (__attribute__((address_space(1))) v4i*)gpv,   // global src
            (__attribute__((address_space(3))) v4i*)lpv,   // LDS dst (addrspacecast)
            /*imm_offset=*/0, /*cpol=*/0);
      }
#else
#pragma unroll
      for (int i = 0; i < 4; ++i) {
        int pid = i * 256 + tid;
        int nc = pid >> 2, q = pid & 3;
        uint4 v = *reinterpret_cast<const uint4*>(wbf + nc * KTOT + kc + q * 8);
        *reinterpret_cast<uint4*>(&Bs[nc * LDA + q * 8]) = v;
      }
#endif
    } else {
      // fallback: convert f32 weights per chunk (no workspace available)
#pragma unroll
      for (int i = 0; i < 16; ++i) {
        int pid = i * 256 + tid;          // 0..4095
        int nc = pid >> 4, kp = pid & 15;
        const float* wp = wgt + nc * KTOT + kc + 2 * kp;
        unsigned int pk = (unsigned int)f32_to_bf16(wp[0])
                        | ((unsigned int)f32_to_bf16(wp[1]) << 16);
        *reinterpret_cast<unsigned int*>(&Bs[nc * LDA + 2 * kp]) = pk;
      }
      if (kc + BK < KTOT) {
        __builtin_prefetch(wgt + (tid >> 4) * KTOT + (kc + BK) + 2 * (tid & 15), 0, 1);
      }
    }

    // ---- stage A: 64x32 sampled values, f32 gather+blend -> bf16 LDS ----
#pragma unroll
    for (int i = 0; i < 8; ++i) {
      int eid = i * 256 + tid;            // 0..2047
      int ml = eid >> 5, kk = eid & 31;
      int k  = kc + kk;
      int c  = k / 9, n = k - c * 9;
      int d  = (ml * 9 + n) * 4;
      int coff = c << 12;                 // c * 64*64
      float v = s_wgt[d+0] * x[s_idx[d+0] + coff]
              + s_wgt[d+1] * x[s_idx[d+1] + coff]
              + s_wgt[d+2] * x[s_idx[d+2] + coff]
              + s_wgt[d+3] * x[s_idx[d+3] + coff];
      As[ml * LDA + kk] = f32_to_bf16(v);
    }

#if HAVE_ASYNC_LDS
    if (WBF) __builtin_amdgcn_s_wait_asynccnt(0);  // this wave's DMA done
#endif
    __syncthreads();                               // all waves' staging visible

    // ---- A fragment: ISA 16-bit A layout (K 0-7/16-23 vs 8-15/24-31 per half) ----
    union { uint4 q[2]; v16bf v; } fa;
    {
      const unsigned short* ap = &As[(mrow + lm) * LDA];
      fa.q[0] = *reinterpret_cast<const uint4*>(ap + 8 * half);
      fa.q[1] = *reinterpret_cast<const uint4*>(ap + 16 + 8 * half);
    }
#pragma unroll
    for (int t = 0; t < 8; ++t) {
      // B fragment: lane = N, halves hold K 0-7/16-23 vs 8-15/24-31 (same as A)
      union { uint4 q[2]; v16bf v; } fb;
      const unsigned short* bp = &Bs[((ngrp + t) * 16 + lm) * LDA];
      fb.q[0] = *reinterpret_cast<const uint4*>(bp + 8 * half);
      fb.q[1] = *reinterpret_cast<const uint4*>(bp + 16 + 8 * half);
      acc[t] = __builtin_amdgcn_wmma_f32_16x16x32_bf16(
          /*neg_a=*/false, fa.v, /*neg_b=*/false, fb.v,
          /*c_mod=*/(short)0, acc[t], /*reuse_a=*/false, /*reuse_b=*/false);
    }
    __syncthreads();   // protect LDS before next chunk's staging
  }

  // ---- 3. write D: VGPR r -> M = r + 8*half, lane lm -> N (ISA C/D layout) ----
#pragma unroll
  for (int t = 0; t < 8; ++t) {
    int ocn = (ngrp + t) * 16 + lm;
#pragma unroll
    for (int r = 0; r < 8; ++r) {
      int m  = m0 + mrow + r + 8 * half;
      int b  = m >> 12;
      int sp = m & 4095;
      out[(b * OC + ocn) * 4096 + sp] = acc[t][r];
    }
  }
}

extern "C" void kernel_launch(void* const* d_in, const int* in_sizes, int n_in,
                              void* d_out, int out_size, void* d_ws, size_t ws_size,
                              hipStream_t stream) {
  (void)in_sizes; (void)n_in; (void)out_size;
  const float* x      = (const float*)d_in[0];   // (4,128,64,64)
  const float* offset = (const float*)d_in[1];   // (4,18,64,64)
  const float* w_conv = (const float*)d_in[2];   // (256,128,3,3)
  float* out = (float*)d_out;                    // (4,256,64,64)

  dim3 grid(16384 / BM);   // 256 workgroups
  dim3 block(256);         // 8 wave32s

  const size_t WBYTES = (size_t)OC * KTOT * sizeof(unsigned short);  // 576 KB
  if (ws_size >= WBYTES) {
    unsigned short* wb = (unsigned short*)d_ws;
    cvt_weights_bf16<<<dim3((OC * KTOT) / 256), dim3(256), 0, stream>>>(w_conv, wb);
    DeformConv2D_wmma_kernel<true><<<grid, block, 0, stream>>>(x, offset, w_conv, wb, out);
  } else {
    DeformConv2D_wmma_kernel<false><<<grid, block, 0, stream>>>(x, offset, w_conv, nullptr, out);
  }
}
